// LSTMSpatial_8349416423884
// MI455X (gfx1250) — compile-verified, matrix operand order checked by reading, not run on hardware
//
#include <hip/hip_runtime.h>
#include <hip/hip_bf16.h>

// ---------------------------------------------------------------------------
// Problem sizes (fixed by the reference)
// ---------------------------------------------------------------------------
#define B_SZ 2048
#define E_SZ 512
#define H_SZ 512
#define D_SZ 512
#define K_SZ 64

// ---------------------------------------------------------------------------
// WMMA types (CDNA5 / gfx1250, wave32)
// ---------------------------------------------------------------------------
typedef __attribute__((ext_vector_type(16))) __bf16 v16bf;
typedef __attribute__((ext_vector_type(8)))  float  v8f;

union Frag16 {
    v16bf v;
    unsigned int u32[8];
};

// one v_cvt_pk_bf16_f32: packs {lo, hi} fp32 -> bf16x2 in a single instruction
// (no clang builtin on this toolchain; the gfx1250 instruction exists — seen in
// compiler output — so emit it directly)
__device__ __forceinline__ unsigned int pack_bf16(float lo, float hi) {
    unsigned int r;
    asm("v_cvt_pk_bf16_f32 %0, %1, %2" : "=v"(r) : "v"(lo), "v"(hi));
    return r;
}

// ---------------------------------------------------------------------------
// Generic  C[M,N] = A1[M,K1]·W1[N,K1]^T (+ A2[M,K2]·W2[N,K2]^T) + bias1 (+bias2)
// Block tile: 128(M) x 128(N), K step 32. 8 waves/block (4 row-groups x 2
// col-groups), each wave a 32x64 tile = 2x4 v_wmma_f32_16x16x32_bf16 frags.
// Register-double-buffered: next K-slab global loads issue before the WMMA
// burst.  LDS rows padded +4 u32 (stride 20 u32 = 80 B): 16B-aligned b128
// fragment loads AND conflict-free banking (20*r mod 64 distinct for r=0..15).
// ---------------------------------------------------------------------------
#define TM 128
#define TN 128
#define TK 32
#define LDW (TK / 2 + 4)   // 20 u32 per LDS row

__global__ __launch_bounds__(256)
void gemm_bias_bf16wmma(const float* __restrict__ A1, int K1,
                        const float* __restrict__ A2, int K2,
                        const float* __restrict__ W1,
                        const float* __restrict__ W2,
                        const float* __restrict__ bias1,
                        const float* __restrict__ bias2,
                        float* __restrict__ C, int N)
{
    __shared__ unsigned int As[TM][LDW];   // packed bf16 pairs, [row][k/2]
    __shared__ unsigned int Bs[TN][LDW];   // [n][k/2]  (== B[k][n] transposed)

    const int tid  = threadIdx.x;
    const int lane = tid & 31;
    const int wave = tid >> 5;        // 0..7
    const int wm   = wave & 3;        // 4 row groups of 32
    const int wn   = wave >> 2;       // 2 col groups of 64
    const int rowBase = blockIdx.y * TM;
    const int colBase = blockIdx.x * TN;
    const int Ktot = K1 + K2;

    v8f acc[2][4];
    #pragma unroll
    for (int i = 0; i < 2; i++)
        #pragma unroll
        for (int j = 0; j < 4; j++)
            #pragma unroll
            for (int e = 0; e < 8; e++)
                acc[i][j][e] = 0.0f;

    const int half = lane >> 4;   // 0: lanes 0-15, 1: lanes 16-31
    const int lidx = lane & 15;

    // per-thread staging registers: 4 float4 for A slab, 4 for B slab
    // tile is [128][32] floats = 1024 float4; idx = i*256+tid;
    // row = idx/8, kq = idx%8 (float4 within row)
    float4 aReg[4], bReg[4];

    auto loadRegs = [&](int k0) {
        #pragma unroll
        for (int i = 0; i < 4; i++) {
            int idx = i * 256 + tid;
            int r   = idx >> 3;
            int gk  = k0 + ((idx & 7) << 2);
            if (gk < K1)
                aReg[i] = *(const float4*)(A1 + (size_t)(rowBase + r) * K1 + gk);
            else
                aReg[i] = *(const float4*)(A2 + (size_t)(rowBase + r) * K2 + (gk - K1));
            if (gk < K1)
                bReg[i] = *(const float4*)(W1 + (size_t)(colBase + r) * K1 + gk);
            else
                bReg[i] = *(const float4*)(W2 + (size_t)(colBase + r) * K2 + (gk - K1));
        }
    };

    loadRegs(0);

    for (int k0 = 0; k0 < Ktot; k0 += TK) {
        // ---- convert (v_cvt_pk_bf16_f32) + stage to LDS as b64 stores ----
        #pragma unroll
        for (int i = 0; i < 4; i++) {
            int idx = i * 256 + tid;
            int r   = idx >> 3;
            int kq  = idx & 7;            // float4 slot -> u32 pair slot kq*2
            uint2 pa = make_uint2(pack_bf16(aReg[i].x, aReg[i].y),
                                  pack_bf16(aReg[i].z, aReg[i].w));
            uint2 pb = make_uint2(pack_bf16(bReg[i].x, bReg[i].y),
                                  pack_bf16(bReg[i].z, bReg[i].w));
            *(uint2*)&As[r][kq * 2] = pa;
            *(uint2*)&Bs[r][kq * 2] = pb;
        }
        __syncthreads();

        // ---- prefetch next K-slab into registers (overlaps WMMA below) ----
        if (k0 + TK < Ktot) loadRegs(k0 + TK);

        // ---- fragments per ISA 7.12.2: two contiguous 4-u32 (16B) groups ----
        Frag16 afrag[2], bfrag[4];
        #pragma unroll
        for (int mi = 0; mi < 2; mi++) {
            int m = wm * 32 + mi * 16 + lidx;
            #pragma unroll
            for (int j = 0; j < 4; j++) {
                afrag[mi].u32[j]     = As[m][4 * half + j];
                afrag[mi].u32[4 + j] = As[m][8 + 4 * half + j];
            }
        }
        #pragma unroll
        for (int ni = 0; ni < 4; ni++) {
            int n = wn * 64 + ni * 16 + lidx;
            #pragma unroll
            for (int j = 0; j < 4; j++) {
                bfrag[ni].u32[j]     = Bs[n][4 * half + j];
                bfrag[ni].u32[4 + j] = Bs[n][8 + 4 * half + j];
            }
        }

        #pragma unroll
        for (int mi = 0; mi < 2; mi++)
            #pragma unroll
            for (int ni = 0; ni < 4; ni++)
                acc[mi][ni] = __builtin_amdgcn_wmma_f32_16x16x32_bf16(
                    /*neg_a=*/false, afrag[mi].v,
                    /*neg_b=*/false, bfrag[ni].v,
                    /*c_mod=*/(short)0, acc[mi][ni],
                    /*reuse_a=*/false, /*reuse_b=*/false);
        __syncthreads();
    }

    // ---- epilogue: C/D layout — lane col = lidx, row = v + 8*half ----
    #pragma unroll
    for (int mi = 0; mi < 2; mi++) {
        #pragma unroll
        for (int ni = 0; ni < 4; ni++) {
            int col = colBase + wn * 64 + ni * 16 + lidx;
            float bsum = bias1[col] + (bias2 ? bias2[col] : 0.0f);
            #pragma unroll
            for (int v = 0; v < 8; v++) {
                int row = rowBase + wm * 32 + mi * 16 + v + 8 * half;
                C[(size_t)row * N + col] = acc[mi][ni][v] + bsum;
            }
        }
    }
}

// ---------------------------------------------------------------------------
// LSTM pointwise: gates [B,4H] (i,f,g,o) + cx -> hy  (c_new is not an output)
// ---------------------------------------------------------------------------
__global__ void lstm_pointwise(const float* __restrict__ gates,
                               const float* __restrict__ cx,
                               float* __restrict__ hy)
{
    int idx = blockIdx.x * blockDim.x + threadIdx.x;
    if (idx >= B_SZ * H_SZ) return;
    int b = idx / H_SZ;
    int j = idx - b * H_SZ;
    const float* g = gates + (size_t)b * 4 * H_SZ;
    float ig = g[j];
    float fg = g[H_SZ + j];
    float gg = g[2 * H_SZ + j];
    float og = g[3 * H_SZ + j];
    float si = 1.0f / (1.0f + __expf(-ig));
    float sf = 1.0f / (1.0f + __expf(-fg));
    float so = 1.0f / (1.0f + __expf(-og));
    float cn = sf * cx[idx] + si * tanhf(gg);
    hy[idx]  = so * tanhf(cn);
}

// ---------------------------------------------------------------------------
// Additive attention: one block per batch row b.
// Stage features[b] (512x64 fp32 = 128KB) into LDS once (HBM read exactly
// once), then: z[k] = sum_d tanh(f+Wh)*aw ; softmax ; cy[d] = sum_k f*w.
// ---------------------------------------------------------------------------
__global__ __launch_bounds__(256)
void attention_kernel(const float* __restrict__ features,
                      const float* __restrict__ Wh,
                      const float* __restrict__ attn_w,
                      const float* __restrict__ attn_b,
                      float* __restrict__ cy)
{
    __shared__ float feat[D_SZ * K_SZ];    // 128 KB
    __shared__ float wh[D_SZ];
    __shared__ float aw[D_SZ];
    __shared__ float zpart[4][K_SZ];
    __shared__ float zf[K_SZ];
    __shared__ float wexp[K_SZ];
    __shared__ float wts[K_SZ];

    const int b   = blockIdx.x;
    const int tid = threadIdx.x;

    // stage features[b] with b128 loads (coalesced), 32 float4 per thread
    const float4* src = (const float4*)(features + (size_t)b * D_SZ * K_SZ);
    float4* dst = (float4*)feat;
    #pragma unroll
    for (int i = 0; i < (D_SZ * K_SZ / 4) / 256; i++)
        dst[i * 256 + tid] = src[i * 256 + tid];
    wh[tid]       = Wh[(size_t)b * D_SZ + tid];
    wh[tid + 256] = Wh[(size_t)b * D_SZ + tid + 256];
    aw[tid]       = attn_w[tid];
    aw[tid + 256] = attn_w[tid + 256];
    __syncthreads();

    // phase 1: partial channel reduction; thread = (k, 128-wide d group)
    {
        int k   = tid & 63;
        int grp = tid >> 6;      // 0..3
        float acc = 0.0f;
        #pragma unroll 4
        for (int dd = 0; dd < 128; dd++) {
            int d = grp * 128 + dd;
            acc += tanhf(feat[d * K_SZ + k] + wh[d]) * aw[d];
        }
        zpart[grp][k] = acc;
    }
    __syncthreads();
    if (tid < K_SZ)
        zf[tid] = zpart[0][tid] + zpart[1][tid] + zpart[2][tid] + zpart[3][tid]
                + attn_b[0];
    __syncthreads();
    if (tid < K_SZ) {
        float mx = -3.4e38f;
        for (int kk = 0; kk < K_SZ; kk++) mx = fmaxf(mx, zf[kk]);
        wexp[tid] = __expf(zf[tid] - mx);
    }
    __syncthreads();
    if (tid < K_SZ) {
        float s = 0.0f;
        for (int kk = 0; kk < K_SZ; kk++) s += wexp[kk];
        wts[tid] = wexp[tid] / s;
    }
    __syncthreads();

    // phase 2: cy[d] = <feat[d,:], w> ; rotate k per lane to avoid 32-way
    // LDS bank conflicts on the stride-64-float rows.
    const int rot = tid & 63;
    for (int d = tid; d < D_SZ; d += 256) {
        float acc = 0.0f;
        #pragma unroll
        for (int kk = 0; kk < K_SZ; kk++) {
            int kx = (kk + rot) & 63;
            acc += feat[d * K_SZ + kx] * wts[kx];
        }
        cy[(size_t)b * D_SZ + d] = acc;
    }
}

// ---------------------------------------------------------------------------
// Launch: gemm(gates) -> pointwise(hy) -> gemm(Wh) -> attention(cy)
// d_out = [ hy (B*H) | cy (B*D) ].  hy doubles as the second GEMM's A operand.
// d_ws  = [ gates (B*4H fp32 = 16MB) | Wh (B*H fp32 = 4MB) ]
// ---------------------------------------------------------------------------
extern "C" void kernel_launch(void* const* d_in, const int* in_sizes, int n_in,
                              void* d_out, int out_size, void* d_ws, size_t ws_size,
                              hipStream_t stream)
{
    const float* inputs   = (const float*)d_in[0];
    const float* hx       = (const float*)d_in[1];
    const float* cx       = (const float*)d_in[2];
    const float* features = (const float*)d_in[3];
    const float* W_ih     = (const float*)d_in[4];
    const float* W_hh     = (const float*)d_in[5];
    const float* b_ih     = (const float*)d_in[6];
    const float* b_hh     = (const float*)d_in[7];
    const float* W_ht     = (const float*)d_in[8];
    const float* b_ht     = (const float*)d_in[9];
    const float* attn_w   = (const float*)d_in[10];
    const float* attn_b   = (const float*)d_in[11];

    float* out   = (float*)d_out;
    float* hy    = out;                           // [B,H]
    float* cy    = out + (size_t)B_SZ * H_SZ;     // [B,D]
    float* gates = (float*)d_ws;                  // [B,4H]
    float* Wh    = (float*)((char*)d_ws + (size_t)B_SZ * 4 * H_SZ * sizeof(float));

    dim3 block(256);

    // 1) gates = [inputs|hx] @ [W_ih|W_hh]^T + (b_ih + b_hh)
    dim3 grid1((4 * H_SZ) / TN, B_SZ / TM);       // 16 x 16
    gemm_bias_bf16wmma<<<grid1, block, 0, stream>>>(
        inputs, E_SZ, hx, H_SZ, W_ih, W_hh, b_ih, b_hh, gates, 4 * H_SZ);

    // 2) LSTM activations -> hy (written straight into d_out)
    lstm_pointwise<<<(B_SZ * H_SZ + 255) / 256, 256, 0, stream>>>(gates, cx, hy);

    // 3) Wh = hy @ W_ht^T + b_ht
    dim3 grid2(H_SZ / TN, B_SZ / TM);             // 4 x 16
    gemm_bias_bf16wmma<<<grid2, block, 0, stream>>>(
        hy, H_SZ, nullptr, 0, W_ht, nullptr, b_ht, nullptr, Wh, H_SZ);

    // 4) attention -> cy
    attention_kernel<<<B_SZ, 256, 0, stream>>>(features, Wh, attn_w, attn_b, cy);
}
